// ProteinEncoder_37400575213655
// MI455X (gfx1250) — compile-verified, matrix operand order checked by reading, not run on hardware
//
#include <hip/hip_runtime.h>
#include <hip/hip_fp16.h>
#include <stdint.h>

// ---------------------------------------------------------------------------
// Model constants (reference: D=256, H=4, DH=64, L=192, B=2, S=1, 4 blocks)
// ---------------------------------------------------------------------------
#define BB 2
#define LL 192
#define DD 256
#define HH 4
#define DHD 64
#define NBLK 4
#define NEGV (-65000.0f)
#define ZROWS (BB * LL * LL)          // 73728 rows of z
#define MROWS (BB * LL)               // 384 rows of m (S==1)

typedef __attribute__((ext_vector_type(16))) _Float16 v16h;
typedef __attribute__((ext_vector_type(8)))  float    v8f;

__device__ __forceinline__ v8f v8f_zero() {
  v8f z; 
#pragma unroll
  for (int i = 0; i < 8; i++) z[i] = 0.0f;
  return z;
}

__device__ __forceinline__ v8f wmma_f16(v16h a, v16h b, v8f c) {
  // D = A(16x32 f16) * B(32x16 f16) + C(16x16 f32)   -> v_wmma_f32_16x16x32_f16
  return __builtin_amdgcn_wmma_f32_16x16x32_f16(false, a, false, b, (short)0, c,
                                                false, false);
}

// Fragment loader for operands stored row-major with K contiguous ([row][k]).
// Works both for A (rows = M) and for B when the B-source is a (N,K) weight
// matrix or an LDS tile stored [n][k].  Per ISA 7.12.2 (16-bit A 16x32):
//   lane = m | (khalf<<4); elems 0..7 -> K in [8*khalf, 8*khalf+8)
//                          elems 8..15 -> K in [16+8*khalf, ...+8)
// i.e. two contiguous 16-byte chunks -> two ds_load_b128.
__device__ __forceinline__ v16h load_frag_rowK(const __half* base, int row,
                                               int ld, int k0, int lane) {
  int r = row + (lane & 15);
  int kh = (lane >> 4) << 3;
  const __half* p = base + (long)r * ld + k0 + kh;
  union { uint4 u[2]; v16h h; } u;
  u.u[0] = *reinterpret_cast<const uint4*>(p);
  u.u[1] = *reinterpret_cast<const uint4*>(p + 16);
  return u.h;
}

__device__ __forceinline__ float gelu_exact(float x) {
  return 0.5f * x * (1.0f + erff(x * 0.70710678118654752f));
}

// ---------------------------------------------------------------------------
// Generic WMMA GEMM:  Out[M,N] = act(A[M,K] @ W[N,K]^T + bias) (+Res)
// A, W are f16; Out fp32 and/or f16.  Block tile 128x128, 8 waves, wave tile
// 32x64 (2x4 fragments of 16x16), K-step 32.
// ---------------------------------------------------------------------------
enum { ACT_NONE = 0, ACT_RELU = 1, ACT_GELU = 2 };

template <int ACT, bool HAS_RES, bool HAS_O32, bool HAS_O16>
__global__ __launch_bounds__(256) void gemm_kernel(
    const __half* __restrict__ A, const __half* __restrict__ W,
    const float* __restrict__ bias, const float* __restrict__ Res,
    float* __restrict__ O32, __half* __restrict__ O16, int M, int N, int K) {
  __shared__ __half As[128 * 32];
  __shared__ __half Bs[128 * 32];
  const int tid = threadIdx.x, lane = tid & 31, wid = tid >> 5;
  const int waveM = wid >> 1;  // 0..3 -> 32-row strip
  const int waveN = wid & 1;   // 0..1 -> 64-col strip
  const long gM0 = (long)blockIdx.y * 128;
  const int gN0 = blockIdx.x * 128;

  v8f acc[2][4];
#pragma unroll
  for (int i = 0; i < 2; i++)
#pragma unroll
    for (int j = 0; j < 4; j++) acc[i][j] = v8f_zero();

  const int r = tid >> 1;
  const int seg = (tid & 1) * 16;
  const uint4 z4 = make_uint4(0u, 0u, 0u, 0u);

  for (int k0 = 0; k0 < K; k0 += 32) {
    const __half* srcA = A + (gM0 + r) * (long)K + k0 + seg;
    *reinterpret_cast<uint4*>(&As[r * 32 + seg]) =
        *reinterpret_cast<const uint4*>(srcA);
    *reinterpret_cast<uint4*>(&As[r * 32 + seg + 8]) =
        *reinterpret_cast<const uint4*>(srcA + 8);
    if (k0 + 64 < K) __builtin_prefetch(srcA + 32, 0, 0);
    const int wrow = gN0 + r;
    if (wrow < N) {
      const __half* srcW = W + (long)wrow * K + k0 + seg;
      *reinterpret_cast<uint4*>(&Bs[r * 32 + seg]) =
          *reinterpret_cast<const uint4*>(srcW);
      *reinterpret_cast<uint4*>(&Bs[r * 32 + seg + 8]) =
          *reinterpret_cast<const uint4*>(srcW + 8);
    } else {
      *reinterpret_cast<uint4*>(&Bs[r * 32 + seg]) = z4;
      *reinterpret_cast<uint4*>(&Bs[r * 32 + seg + 8]) = z4;
    }
    __syncthreads();
    v16h a0 = load_frag_rowK(As, waveM * 32, 32, 0, lane);
    v16h a1 = load_frag_rowK(As, waveM * 32 + 16, 32, 0, lane);
#pragma unroll
    for (int ni = 0; ni < 4; ni++) {
      v16h bb = load_frag_rowK(Bs, waveN * 64 + ni * 16, 32, 0, lane);
      acc[0][ni] = wmma_f16(a0, bb, acc[0][ni]);
      acc[1][ni] = wmma_f16(a1, bb, acc[1][ni]);
    }
    __syncthreads();
  }

#pragma unroll
  for (int mi = 0; mi < 2; mi++)
#pragma unroll
    for (int ni = 0; ni < 4; ni++) {
      int colL = waveN * 64 + ni * 16 + (lane & 15);
      int gN = gN0 + colL;
      if (gN < N) {
        float bv = bias[gN];
#pragma unroll
        for (int rr = 0; rr < 8; rr++) {
          int rowL = waveM * 32 + mi * 16 + ((lane >> 4) << 3) + rr;
          long idx = (gM0 + rowL) * (long)N + gN;
          float v = acc[mi][ni][rr] + bv;
          if (ACT == ACT_RELU) v = fmaxf(v, 0.0f);
          if (ACT == ACT_GELU) v = gelu_exact(v);
          if (HAS_RES) v += Res[idx];
          if (HAS_O32) O32[idx] = v;
          if (HAS_O16) O16[idx] = __float2half(v);
        }
      }
    }
}

// ---------------------------------------------------------------------------
// Fused transition MLP:  Out += linear2(gelu(linear1(X)))
// X: (M,256) f16, W1: (1024,256) f16, W2: (256,1024) f16, Out: (M,256) fp32
// 64 rows per workgroup; 64x1024 hidden kept in LDS (never hits HBM).
// Dynamic LDS: Xs 32KB + Hs 128KB + Wt 8KB = 172032 B (<320KB/WGP).
// ---------------------------------------------------------------------------
__global__ __launch_bounds__(256) void mlp_fused_kernel(
    const __half* __restrict__ X, const __half* __restrict__ W1,
    const float* __restrict__ b1, const __half* __restrict__ W2,
    const float* __restrict__ b2, float* __restrict__ Out) {
  extern __shared__ char smem[];
  __half* Xs = reinterpret_cast<__half*>(smem);  // 64*256
  __half* Hs = Xs + 64 * 256;                    // 64*1024
  __half* Wt = Hs + 64 * 1024;                   // 128*32
  __half* Bt = Xs;                               // phase2 alias (256*32 fits)

  const int tid = threadIdx.x, lane = tid & 31, wid = tid >> 5;
  const long gRow0 = (long)blockIdx.x * 64;
  const int wM = wid >> 2;   // 0..1 -> 32-row strip
  const int wN = wid & 3;    // 0..3

  for (int idx = tid; idx < 2048; idx += 256) {
    int rr = idx >> 5, sg = (idx & 31) * 8;
    *reinterpret_cast<uint4*>(&Xs[rr * 256 + sg]) =
        *reinterpret_cast<const uint4*>(X + (gRow0 + rr) * 256 + sg);
  }
  __syncthreads();

  // ---- phase 1: H = gelu(X @ W1^T + b1), in 8 chunks of 128 hidden cols ----
  for (int ch = 0; ch < 8; ch++) {
    const int nh0 = ch * 128;
    v8f acc[2][2];
#pragma unroll
    for (int i = 0; i < 2; i++)
#pragma unroll
      for (int j = 0; j < 2; j++) acc[i][j] = v8f_zero();
    for (int k0 = 0; k0 < 256; k0 += 32) {
      int rr = tid >> 1, sg = (tid & 1) * 16;
      const __half* src = W1 + (long)(nh0 + rr) * 256 + k0 + sg;
      *reinterpret_cast<uint4*>(&Wt[rr * 32 + sg]) =
          *reinterpret_cast<const uint4*>(src);
      *reinterpret_cast<uint4*>(&Wt[rr * 32 + sg + 8]) =
          *reinterpret_cast<const uint4*>(src + 8);
      __syncthreads();
      v16h a0 = load_frag_rowK(Xs, wM * 32, 256, k0, lane);
      v16h a1 = load_frag_rowK(Xs, wM * 32 + 16, 256, k0, lane);
#pragma unroll
      for (int ni = 0; ni < 2; ni++) {
        v16h bb = load_frag_rowK(Wt, wN * 32 + ni * 16, 32, 0, lane);
        acc[0][ni] = wmma_f16(a0, bb, acc[0][ni]);
        acc[1][ni] = wmma_f16(a1, bb, acc[1][ni]);
      }
      __syncthreads();
    }
#pragma unroll
    for (int mi = 0; mi < 2; mi++)
#pragma unroll
      for (int ni = 0; ni < 2; ni++) {
        int nh = nh0 + wN * 32 + ni * 16 + (lane & 15);
        float bv = b1[nh];
#pragma unroll
        for (int rr = 0; rr < 8; rr++) {
          int row = wM * 32 + mi * 16 + ((lane >> 4) << 3) + rr;
          float hv = gelu_exact(acc[mi][ni][rr] + bv);
          Hs[row * 1024 + nh] = __float2half(hv);
        }
      }
  }
  __syncthreads();

  // ---- phase 2: Out += H @ W2^T + b2 ----
  v8f acc2[2][4];
#pragma unroll
  for (int i = 0; i < 2; i++)
#pragma unroll
    for (int j = 0; j < 4; j++) acc2[i][j] = v8f_zero();
  for (int k0 = 0; k0 < 1024; k0 += 32) {
    const __half* src = W2 + (long)tid * 1024 + k0;
    *reinterpret_cast<uint4*>(&Bt[tid * 32]) =
        *reinterpret_cast<const uint4*>(src);
    *reinterpret_cast<uint4*>(&Bt[tid * 32 + 8]) =
        *reinterpret_cast<const uint4*>(src + 8);
    *reinterpret_cast<uint4*>(&Bt[tid * 32 + 16]) =
        *reinterpret_cast<const uint4*>(src + 16);
    *reinterpret_cast<uint4*>(&Bt[tid * 32 + 24]) =
        *reinterpret_cast<const uint4*>(src + 24);
    __syncthreads();
    v16h a0 = load_frag_rowK(Hs, wM * 32, 1024, k0, lane);
    v16h a1 = load_frag_rowK(Hs, wM * 32 + 16, 1024, k0, lane);
#pragma unroll
    for (int ni = 0; ni < 4; ni++) {
      v16h bb = load_frag_rowK(Bt, wN * 64 + ni * 16, 32, 0, lane);
      acc2[0][ni] = wmma_f16(a0, bb, acc2[0][ni]);
      acc2[1][ni] = wmma_f16(a1, bb, acc2[1][ni]);
    }
    __syncthreads();
  }
#pragma unroll
  for (int mi = 0; mi < 2; mi++)
#pragma unroll
    for (int ni = 0; ni < 4; ni++) {
      int n = wN * 64 + ni * 16 + (lane & 15);
      float bv = b2[n];
#pragma unroll
      for (int rr = 0; rr < 8; rr++) {
        int row = wM * 32 + mi * 16 + ((lane >> 4) << 3) + rr;
        long idx = (gRow0 + row) * 256 + n;
        Out[idx] = acc2[mi][ni][rr] + bv + Out[idx];
      }
    }
}

// ---------------------------------------------------------------------------
// Fused attention for one (b, i, h): S = Q K^T * 0.125 (masked) -> softmax ->
// O = P V.  Q/K/V f16 in (rows, 256) layout, head slice at h*64.
// Dynamic LDS: Qs/Ks/Vst (3*24576B) + Ss f32 (147456B) + Ph f16 (73728B)
//            = 294912 B.
// ---------------------------------------------------------------------------
__global__ __launch_bounds__(256) void attention_kernel(
    const __half* __restrict__ Q, const __half* __restrict__ Kt,
    const __half* __restrict__ V, const float* __restrict__ maskK,
    const float* __restrict__ maskQ, __half* __restrict__ O, int I,
    long mStrideB, long mStrideI) {
  extern __shared__ char smem[];
  __half* Qs = reinterpret_cast<__half*>(smem);     // [192][64]
  __half* Ks = Qs + 192 * 64;                       // [192][64]
  __half* Vst = Ks + 192 * 64;                      // [64][192] (transposed)
  float* Ss = reinterpret_cast<float*>(Vst + 192 * 64);  // [192][192]
  __half* Ph = reinterpret_cast<__half*>(Ss + 192 * 192); // [192][192]

  const int tid = threadIdx.x, lane = tid & 31, wid = tid >> 5;
  const int gid = blockIdx.x;
  const int h = gid & 3;
  const int rest = gid >> 2;
  const int i = rest % I;
  const int b = rest / I;
  const long base = ((long)(b * I + i) * 192) * 256 + h * 64;
  const float* mk = maskK + (long)b * mStrideB + (long)i * mStrideI;
  const float* mq = maskQ ? (maskQ + (long)b * 192) : nullptr;

  for (int idx = tid; idx < 1536; idx += 256) {
    int j = idx >> 3, sg = (idx & 7) * 8;
    *reinterpret_cast<uint4*>(&Qs[j * 64 + sg]) =
        *reinterpret_cast<const uint4*>(Q + base + (long)j * 256 + sg);
    *reinterpret_cast<uint4*>(&Ks[j * 64 + sg]) =
        *reinterpret_cast<const uint4*>(Kt + base + (long)j * 256 + sg);
  }
  for (int idx = tid; idx < 12288; idx += 256) {
    int j = idx >> 6, d = idx & 63;
    Vst[d * 192 + j] = V[base + (long)j * 256 + d];
  }
  __syncthreads();

  // ---- scores:  144 16x16 tiles, 18 per wave, K=64 in two WMMA steps ----
  for (int t = wid; t < 144; t += 8) {
    int ti = t / 12, tj = t % 12;
    v8f acc = v8f_zero();
    acc = wmma_f16(load_frag_rowK(Qs, ti * 16, 64, 0, lane),
                   load_frag_rowK(Ks, tj * 16, 64, 0, lane), acc);
    acc = wmma_f16(load_frag_rowK(Qs, ti * 16, 64, 32, lane),
                   load_frag_rowK(Ks, tj * 16, 64, 32, lane), acc);
#pragma unroll
    for (int rr = 0; rr < 8; rr++) {
      int row = ti * 16 + ((lane >> 4) << 3) + rr;
      Ss[row * 192 + tj * 16 + (lane & 15)] = acc[rr];
    }
  }
  __syncthreads();

  // ---- masked softmax: wave handles 24 rows, lane handles 6 cols ----
  for (int jr = 0; jr < 24; jr++) {
    int j = wid * 24 + jr;
    bool qg = mq ? (mq[j] > 0.0f) : true;
    float v[6];
    float mx = -3.0e38f;
#pragma unroll
    for (int s6 = 0; s6 < 6; s6++) {
      int c = lane + s6 * 32;
      float sv = Ss[j * 192 + c] * 0.125f;   // DH^-0.5
      sv = (qg && mk[c] > 0.0f) ? sv : NEGV;
      v[s6] = sv;
      mx = fmaxf(mx, sv);
    }
#pragma unroll
    for (int m = 16; m > 0; m >>= 1) mx = fmaxf(mx, __shfl_xor(mx, m, 32));
    float sum = 0.0f;
#pragma unroll
    for (int s6 = 0; s6 < 6; s6++) {
      v[s6] = expf(v[s6] - mx);
      sum += v[s6];
    }
#pragma unroll
    for (int m = 16; m > 0; m >>= 1) sum += __shfl_xor(sum, m, 32);
    float inv = 1.0f / sum;
#pragma unroll
    for (int s6 = 0; s6 < 6; s6++)
      Ph[j * 192 + lane + s6 * 32] = __float2half(v[s6] * inv);
  }
  __syncthreads();

  // ---- O = P @ V: 48 tiles (12 x 4), 6 per wave, K=192 in 6 WMMA steps ----
  for (int t = wid; t < 48; t += 8) {
    int ti = t >> 2, tj = t & 3;
    v8f acc = v8f_zero();
#pragma unroll
    for (int ks = 0; ks < 6; ks++)
      acc = wmma_f16(load_frag_rowK(Ph, ti * 16, 192, ks * 32, lane),
                     load_frag_rowK(Vst, tj * 16, 192, ks * 32, lane), acc);
#pragma unroll
    for (int rr = 0; rr < 8; rr++) {
      int j = ti * 16 + ((lane >> 4) << 3) + rr;
      int d = tj * 16 + (lane & 15);
      O[base + (long)j * 256 + d] = __float2half(acc[rr]);
    }
  }
}

// ---------------------------------------------------------------------------
// Elementwise / reduction helper kernels
// ---------------------------------------------------------------------------
__global__ __launch_bounds__(256) void layernorm_kernel(
    const float* __restrict__ X, const float* __restrict__ g,
    const float* __restrict__ bt, __half* __restrict__ O) {
  __shared__ float red[8];
  const int row = blockIdx.x, tid = threadIdx.x, lane = tid & 31,
            wid = tid >> 5;
  float x = X[(long)row * 256 + tid];
  float s = x;
#pragma unroll
  for (int m = 16; m > 0; m >>= 1) s += __shfl_xor(s, m, 32);
  if (lane == 0) red[wid] = s;
  __syncthreads();
  float tot = 0.0f;
#pragma unroll
  for (int k = 0; k < 8; k++) tot += red[k];
  float mu = tot * (1.0f / 256.0f);
  float d = x - mu;
  float s2 = d * d;
#pragma unroll
  for (int m = 16; m > 0; m >>= 1) s2 += __shfl_xor(s2, m, 32);
  __syncthreads();
  if (lane == 0) red[wid] = s2;
  __syncthreads();
  float var = 0.0f;
#pragma unroll
  for (int k = 0; k < 8; k++) var += red[k];
  var *= (1.0f / 256.0f);
  float y = d * rsqrtf(var + 1e-5f) * g[tid] + bt[tid];
  O[(long)row * 256 + tid] = __float2half(y);
}

__global__ void f2h_kernel(const float* x, __half* o, int n) {
  int i = blockIdx.x * 256 + threadIdx.x;
  if (i < n) o[i] = __float2half(x[i]);
}

__global__ void add2h_kernel(const float* a, const float* b, __half* o, int n) {
  int i = blockIdx.x * 256 + threadIdx.x;
  if (i < n) o[i] = __float2half(a[i] + b[i]);
}

__global__ void embed_m_kernel(const int* seq, const float* emb, float* M,
                               float* sm) {
  int row = blockIdx.x, c = threadIdx.x;
  int s = seq[row];
  M[(long)row * 256 + c] = emb[s * 256 + c];
  if (c == 0) sm[row] = (s > 0) ? 1.0f : 0.0f;
}

__global__ void embed_z_kernel(const float* coords, float* Z) {
  int x = blockIdx.x, c = threadIdx.x;
  int j = x % 192, i = (x / 192) % 192, b = x / (192 * 192);
  const float* ci = coords + ((long)b * 192 + i) * 3;
  const float* cj = coords + ((long)b * 192 + j) * 3;
  float dx = ci[0] - cj[0], dy = ci[1] - cj[1], dz = ci[2] - cj[2];
  float sq = dx * dx + dy * dy + dz * dz;
  float dist = (sq > 0.0f) ? sqrtf(sq) : 0.0f;
  float v = 0.0f;
  if (c < 10) v = sinf(dist * exp2f((float)c));
  Z[(long)x * 256 + c] = v;
}

__global__ void pairmask_kernel(const float* sm, float* pm) {
  int idx = blockIdx.x * 256 + threadIdx.x;
  if (idx >= ZROWS) return;
  int j = idx % 192, i = (idx / 192) % 192, b = idx / (192 * 192);
  pm[idx] = sm[b * 192 + i] * sm[b * 192 + j];
}

__global__ void meanj_kernel(const __half* Z, __half* Out) {
  int row = blockIdx.x, c = threadIdx.x;
  const __half* p = Z + (long)row * 192 * 256 + c;
  float s = 0.0f;
  for (int j = 0; j < 192; j++) s += __half2float(p[(long)j * 256]);
  Out[(long)row * 256 + c] = __float2half(s * (1.0f / 192.0f));
}

__global__ void bcast_add_kernel(float* Z, const float* P) {
  long idx = blockIdx.x;
  int c = threadIdx.x;
  long bi = idx / 192;
  Z[idx * 256 + c] += P[bi * 256 + c];
}

// ---- threefry2x32 (matches jax.random with key(42)) + final sampling ------
__device__ __forceinline__ uint32_t rotl32(uint32_t x, int d) {
  return (x << d) | (x >> (32 - d));
}
__device__ void threefry2x32(uint32_t k0, uint32_t k1, uint32_t x0, uint32_t x1,
                             uint32_t* o0, uint32_t* o1) {
  uint32_t ks[3] = {k0, k1, k0 ^ k1 ^ 0x1BD11BDAu};
  const int R[8] = {13, 15, 26, 6, 17, 29, 16, 24};
  x0 += ks[0];
  x1 += ks[1];
  for (int i = 0; i < 5; i++) {
    for (int r = 0; r < 4; r++) {
      x0 += x1;
      x1 = rotl32(x1, R[(i & 1) * 4 + r]);
      x1 ^= x0;
    }
    x0 += ks[(i + 1) % 3];
    x1 += ks[(i + 2) % 3] + (uint32_t)(i + 1);
  }
  *o0 = x0;
  *o1 = x1;
}

__global__ void latent_kernel(const float* lat, float* out) {
  int i = blockIdx.x * 256 + threadIdx.x;  // 0 .. 7679  (B*L*LATENT)
  if (i >= 7680) return;
  int row = i / 20, c = i % 20;
  float mu = lat[row * 40 + c];
  float lv = lat[row * 40 + 20 + c];
  lv = fminf(20.0f, fmaxf(-20.0f, lv));
  float sd = expf(0.5f * lv);
  uint32_t x0 = (i < 3840) ? (uint32_t)i : (uint32_t)(i - 3840);
  uint32_t x1 = (i < 3840) ? (uint32_t)(i + 3840) : (uint32_t)i;
  uint32_t o0, o1;
  threefry2x32(0u, 42u, x0, x1, &o0, &o1);
  uint32_t bits = (i < 3840) ? o0 : o1;
  float u01 = __uint_as_float((bits >> 9) | 0x3f800000u) - 1.0f;
  const float lo = -0.99999994f;
  float val = fmaxf(lo, u01 * (1.0f - lo) + lo);
  float eps = 1.41421356237f * erfinvf(val);
  out[i] = mu + eps * sd;
  out[7680 + i] = mu;
  out[15360 + i] = lv;
}

// ---------------------------------------------------------------------------
// Host orchestration
// ---------------------------------------------------------------------------
struct LinW { const float* w; const float* b; };
struct LnW  { const float* g; const float* b; };
struct BlockW {
  LinW msa_q, msa_k, msa_v, msa_o, msa_t1, msa_t2;
  LinW pa_q, pa_k, pa_v, pa_o, pair_t1, pair_t2;
  LinW mpa_msa, mpa_out, pmu_pair, pmu_msa, pmu_out;
  LnW mn1, mn2, mn3, pn1, pn2, pn3;
};
struct BlockW16 {
  __half *msa_q, *msa_k, *msa_v, *msa_o, *msa_t1, *msa_t2;
  __half *pa_q, *pa_k, *pa_v, *pa_o, *pair_t1, *pair_t2;
  __half *mpa_msa, *mpa_out, *pmu_pair, *pmu_msa, *pmu_out;
};

template <int ACT, bool RES, bool O32, bool O16>
static void launch_gemm(const __half* A, const __half* W, const float* bias,
                        const float* Res, float* o32, __half* o16, int M, int N,
                        int K, hipStream_t st) {
  dim3 grid((N + 127) / 128, M / 128);
  gemm_kernel<ACT, RES, O32, O16>
      <<<grid, 256, 0, st>>>(A, W, bias, Res, o32, o16, M, N, K);
}

extern "C" void kernel_launch(void* const* d_in, const int* in_sizes, int n_in,
                              void* d_out, int out_size, void* d_ws,
                              size_t ws_size, hipStream_t stream) {
  (void)in_sizes; (void)out_size; (void)ws_size;
  if (n_in < 191) return;  // expected flattened-leaf count

  // ---- parse inputs (insertion order of setup_inputs dict/pytree) ----
  int c = 0;
  const int* seq = (const int*)d_in[c++];
  const float* coords = (const float*)d_in[c++];
  const float* seq_emb = (const float*)d_in[c++];
  BlockW blk[NBLK];
  auto lin = [&](LinW& L) {
    L.w = (const float*)d_in[c++];
    L.b = (const float*)d_in[c++];
  };
  auto lnp = [&](LnW& L) {
    L.g = (const float*)d_in[c++];
    L.b = (const float*)d_in[c++];
  };
  for (int bI = 0; bI < NBLK; bI++) {
    BlockW& B = blk[bI];
    lin(B.msa_q); lin(B.msa_k); lin(B.msa_v); lin(B.msa_o);
    lin(B.msa_t1); lin(B.msa_t2);
    lin(B.pa_q); lin(B.pa_k); lin(B.pa_v); lin(B.pa_o);
    lin(B.pair_t1); lin(B.pair_t2);
    lin(B.mpa_msa); lin(B.mpa_out);
    lin(B.pmu_pair); lin(B.pmu_msa); lin(B.pmu_out);
    lnp(B.mn1); lnp(B.mn2); lnp(B.mn3);
    lnp(B.pn1); lnp(B.pn2); lnp(B.pn3);
  }
  LinW lp1, lp2;
  lin(lp1); lin(lp2);

  // ---- workspace bump allocator ----
  char* wsp = (char*)d_ws;
  size_t cur = 0;
  auto alloc = [&](size_t bytes) -> void* {
    cur = (cur + 255) & ~(size_t)255;
    void* p = wsp + cur;
    cur += bytes;
    return p;
  };
  auto conv = [&](const float* src, size_t n) -> __half* {
    __half* dst = (__half*)alloc(n * sizeof(__half));
    f2h_kernel<<<(int)((n + 255) / 256), 256, 0, stream>>>(src, dst, (int)n);
    return dst;
  };

  // f16 weight copies (made once per launch; deterministic)
  BlockW16 w16[NBLK];
  for (int bI = 0; bI < NBLK; bI++) {
    BlockW& B = blk[bI];
    BlockW16& W = w16[bI];
    const size_t DxD = (size_t)DD * DD, D4 = (size_t)4 * DD * DD;
    W.msa_q = conv(B.msa_q.w, DxD);  W.msa_k = conv(B.msa_k.w, DxD);
    W.msa_v = conv(B.msa_v.w, DxD);  W.msa_o = conv(B.msa_o.w, DxD);
    W.msa_t1 = conv(B.msa_t1.w, D4); W.msa_t2 = conv(B.msa_t2.w, D4);
    W.pa_q = conv(B.pa_q.w, DxD);    W.pa_k = conv(B.pa_k.w, DxD);
    W.pa_v = conv(B.pa_v.w, DxD);    W.pa_o = conv(B.pa_o.w, DxD);
    W.pair_t1 = conv(B.pair_t1.w, D4); W.pair_t2 = conv(B.pair_t2.w, D4);
    W.mpa_msa = conv(B.mpa_msa.w, DxD); W.mpa_out = conv(B.mpa_out.w, DxD);
    W.pmu_pair = conv(B.pmu_pair.w, DxD); W.pmu_msa = conv(B.pmu_msa.w, DxD);
    W.pmu_out = conv(B.pmu_out.w, DxD);
  }
  __half* lp1w = conv(lp1.w, (size_t)DD * DD);
  __half* lp2w = conv(lp2.w, (size_t)40 * DD);

  // activation buffers
  float* zbuf = (float*)alloc((size_t)ZROWS * DD * 4);        // z (fp32)
  float* mbuf = (float*)alloc((size_t)MROWS * DD * 4);        // m (fp32)
  float* smbuf = (float*)alloc((size_t)MROWS * 4);            // seq mask
  float* pmbuf = (float*)alloc((size_t)ZROWS * 4);            // pair mask
  __half* ln16 = (__half*)alloc((size_t)ZROWS * DD * 2);      // LN output f16
  __half* q16 = (__half*)alloc((size_t)ZROWS * DD * 2);
  __half* k16 = (__half*)alloc((size_t)ZROWS * DD * 2);
  __half* v16 = (__half*)alloc((size_t)ZROWS * DD * 2);
  __half* ao16 = (__half*)alloc((size_t)ZROWS * DD * 2);      // attn out f16
  float* proj = (float*)alloc((size_t)MROWS * DD * 4);
  float* ctx = (float*)alloc((size_t)MROWS * DD * 4);
  __half* zmean16 = (__half*)alloc((size_t)MROWS * DD * 2);
  __half* m16 = (__half*)alloc((size_t)MROWS * DD * 2);
  __half* sum16 = (__half*)alloc((size_t)MROWS * DD * 2);
  __half* lath16 = (__half*)alloc((size_t)MROWS * DD * 2);
  float* latbuf = (float*)alloc((size_t)MROWS * 40 * 4);

  const size_t ATTN_LDS = 3 * 192 * 64 * 2 + 192 * 192 * 4 + 192 * 192 * 2;
  const size_t MLP_LDS = (64 * 256 + 64 * 1024 + 128 * 32) * 2;
  const int NM = MROWS * DD;

  // ---- embeddings / masks ----
  embed_m_kernel<<<MROWS, 256, 0, stream>>>(seq, seq_emb, mbuf, smbuf);
  embed_z_kernel<<<ZROWS, 256, 0, stream>>>(coords, zbuf);
  pairmask_kernel<<<(ZROWS + 255) / 256, 256, 0, stream>>>(smbuf, pmbuf);

  // ---- evoformer blocks ----
  for (int bI = 0; bI < NBLK; bI++) {
    BlockW& B = blk[bI];
    BlockW16& W = w16[bI];

    // -- MSA self-attention (m is 384 x 256) --
    layernorm_kernel<<<MROWS, 256, 0, stream>>>(mbuf, B.mn1.g, B.mn1.b, ln16);
    launch_gemm<ACT_NONE, false, false, true>(ln16, W.msa_q, B.msa_q.b, nullptr,
                                              nullptr, q16, MROWS, DD, DD, stream);
    launch_gemm<ACT_NONE, false, false, true>(ln16, W.msa_k, B.msa_k.b, nullptr,
                                              nullptr, k16, MROWS, DD, DD, stream);
    launch_gemm<ACT_NONE, false, false, true>(ln16, W.msa_v, B.msa_v.b, nullptr,
                                              nullptr, v16, MROWS, DD, DD, stream);
    attention_kernel<<<BB * 1 * HH, 256, ATTN_LDS, stream>>>(
        q16, k16, v16, smbuf, nullptr, ao16, 1, LL, 0);
    launch_gemm<ACT_NONE, true, true, false>(ao16, W.msa_o, B.msa_o.b, mbuf,
                                             mbuf, nullptr, MROWS, DD, DD, stream);
    // -- MSA transition (fused MLP, residual in-place) --
    layernorm_kernel<<<MROWS, 256, 0, stream>>>(mbuf, B.mn2.g, B.mn2.b, ln16);
    mlp_fused_kernel<<<MROWS / 64, 256, MLP_LDS, stream>>>(
        ln16, W.msa_t1, B.msa_t1.b, W.msa_t2, B.msa_t2.b, mbuf);

    // -- MSA->pair attention (q used as q/k/v), broadcast into z --
    layernorm_kernel<<<MROWS, 256, 0, stream>>>(mbuf, B.mn3.g, B.mn3.b, ln16);
    launch_gemm<ACT_NONE, false, false, true>(ln16, W.mpa_msa, B.mpa_msa.b,
                                              nullptr, nullptr, q16, MROWS, DD,
                                              DD, stream);
    attention_kernel<<<BB * 1 * HH, 256, ATTN_LDS, stream>>>(
        q16, q16, q16, smbuf, smbuf, ao16, 1, LL, 0);
    launch_gemm<ACT_NONE, false, true, false>(ao16, W.mpa_out, B.mpa_out.b,
                                              nullptr, proj, nullptr, MROWS, DD,
                                              DD, stream);
    bcast_add_kernel<<<ZROWS, 256, 0, stream>>>(zbuf, proj);  // z_update

    // -- pair self-attention over j (73728 rows) --
    layernorm_kernel<<<ZROWS, 256, 0, stream>>>(zbuf, B.pn1.g, B.pn1.b, ln16);
    launch_gemm<ACT_NONE, false, false, true>(ln16, W.pa_q, B.pa_q.b, nullptr,
                                              nullptr, q16, ZROWS, DD, DD, stream);
    launch_gemm<ACT_NONE, false, false, true>(ln16, W.pa_k, B.pa_k.b, nullptr,
                                              nullptr, k16, ZROWS, DD, DD, stream);
    launch_gemm<ACT_NONE, false, false, true>(ln16, W.pa_v, B.pa_v.b, nullptr,
                                              nullptr, v16, ZROWS, DD, DD, stream);
    attention_kernel<<<BB * LL * HH, 256, ATTN_LDS, stream>>>(
        q16, k16, v16, pmbuf, nullptr, ao16, LL, (long)LL * LL, LL);
    launch_gemm<ACT_NONE, true, true, false>(ao16, W.pa_o, B.pa_o.b, zbuf, zbuf,
                                             nullptr, ZROWS, DD, DD, stream);
    // -- pair transition (fused MLP, in-place) --
    layernorm_kernel<<<ZROWS, 256, 0, stream>>>(zbuf, B.pn2.g, B.pn2.b, ln16);
    mlp_fused_kernel<<<ZROWS / 64, 256, MLP_LDS, stream>>>(
        ln16, W.pair_t1, B.pair_t1.b, W.pair_t2, B.pair_t2.b, zbuf);

    // -- pair->MSA update: mean_j(linear(z_ln)) == linear(mean_j(z_ln)) --
    layernorm_kernel<<<ZROWS, 256, 0, stream>>>(zbuf, B.pn3.g, B.pn3.b, ln16);
    meanj_kernel<<<MROWS, 256, 0, stream>>>(ln16, zmean16);
    launch_gemm<ACT_NONE, false, true, false>(zmean16, W.pmu_pair, B.pmu_pair.b,
                                              nullptr, ctx, nullptr, MROWS, DD,
                                              DD, stream);
    f2h_kernel<<<(NM + 255) / 256, 256, 0, stream>>>(mbuf, m16, NM);
    launch_gemm<ACT_NONE, false, true, false>(m16, W.pmu_msa, B.pmu_msa.b,
                                              nullptr, proj, nullptr, MROWS, DD,
                                              DD, stream);
    add2h_kernel<<<(NM + 255) / 256, 256, 0, stream>>>(proj, ctx, sum16, NM);
    launch_gemm<ACT_NONE, true, true, false>(sum16, W.pmu_out, B.pmu_out.b,
                                             mbuf, mbuf, nullptr, MROWS, DD, DD,
                                             stream);
  }

  // ---- latent head: relu(lp1(m)) -> lp2 -> reparameterize (threefry) ----
  f2h_kernel<<<(NM + 255) / 256, 256, 0, stream>>>(mbuf, m16, NM);
  launch_gemm<ACT_RELU, false, false, true>(m16, lp1w, lp1.b, nullptr, nullptr,
                                            lath16, MROWS, DD, DD, stream);
  launch_gemm<ACT_NONE, false, true, false>(lath16, lp2w, lp2.b, nullptr,
                                            latbuf, nullptr, MROWS, 40, DD,
                                            stream);
  latent_kernel<<<(7680 + 255) / 256, 256, 0, stream>>>(latbuf, (float*)d_out);
}
